// KnnModule_71012989272143
// MI455X (gfx1250) — matrix operand order repaired and checked by reading, not compiled, not run on hardware
//
#include <hip/hip_runtime.h>
#include <math.h>

#define B_Q      256
#define N_TRAIN  100000
#define D_FEAT   512
#define MAXK     200
#define NCLS     1000
#define INV_TEMP (1.0f / 0.07f)
#define LDS_STRIDE 520   // 512 bf16 + 8 pad -> conflict-free b128 LDS reads

typedef __attribute__((ext_vector_type(16))) __bf16 v16bf;
typedef __attribute__((ext_vector_type(8)))  __bf16 v8bf;
typedef __attribute__((ext_vector_type(8)))  float  v8f;
typedef __attribute__((ext_vector_type(4)))  int    v4i;
typedef __attribute__((address_space(1))) v4i gv4i;   // global int4
typedef __attribute__((address_space(3))) v4i lv4i;   // LDS int4

__device__ __forceinline__ float4 load4(const float* p) {
  return *reinterpret_cast<const float4*>(p);
}

// ---------------------------------------------------------------------------
// Phase 0: streaming f32 -> bf16 convert (train bank + query features)
// ---------------------------------------------------------------------------
__global__ void __launch_bounds__(256) cvt_f32_to_bf16(const float* __restrict__ src,
                                                       v8bf* __restrict__ dst, int n8) {
  const int i = blockIdx.x * 256 + threadIdx.x;
  if (i >= n8) return;
  const float4 a = load4(src + (size_t)i * 8);
  const float4 b = load4(src + (size_t)i * 8 + 4);
  v8bf o;
  o[0] = (__bf16)a.x; o[1] = (__bf16)a.y; o[2] = (__bf16)a.z; o[3] = (__bf16)a.w;
  o[4] = (__bf16)b.x; o[5] = (__bf16)b.y; o[6] = (__bf16)b.z; o[7] = (__bf16)b.w;
  dst[i] = o;
}

// ---------------------------------------------------------------------------
// Phase 1: sims = A (256x512) @ train^T, bf16 WMMA, f32 accum.
// Block = 8 waves. Block tile: full M=256 x 32 N columns (train read once).
// B tile (32x512 bf16 = 32KB) staged into LDS with gfx1250 async loads.
// Each wave: 32x32 macro-tile = 4 WMMAs per k-step, 1KB loaded per WMMA.
// ---------------------------------------------------------------------------
__global__ void __launch_bounds__(256) knn_gemm_bf16(const __bf16* __restrict__ A16,
                                                     const __bf16* __restrict__ B16,
                                                     float* __restrict__ sims) {
  __shared__ __bf16 bt[32 * LDS_STRIDE];

  const int tid   = threadIdx.x;
  const int nbase = blockIdx.x * 32;

  // ---- stage 32x512 bf16 B tile into LDS (async path if available) ----
  {
    const int r = tid >> 3;            // tile row (N)
    const int c = (tid & 7) * 64;      // 64 bf16 = 128B per thread
    const __bf16* g = B16 + (size_t)(nbase + r) * D_FEAT + c;
    __bf16* l = &bt[r * LDS_STRIDE + c];
#pragma unroll
    for (int u = 0; u < 8; ++u) {
#if __has_builtin(__builtin_amdgcn_global_load_async_to_lds_b128)
      __builtin_amdgcn_global_load_async_to_lds_b128(
          (gv4i*)(g + u * 8), (lv4i*)(l + u * 8), 0, 0);
#else
      *(uint4*)(l + u * 8) = *(const uint4*)(g + u * 8);
#endif
    }
  }
#if __has_builtin(__builtin_amdgcn_s_wait_asynccnt)
  __builtin_amdgcn_s_wait_asynccnt(0);
#endif
  __syncthreads();

  const int wave  = tid >> 5;
  const int lane  = tid & 31;
  const int lm    = lane & 15;
  const int lh    = lane >> 4;
  const int mbase = wave * 32;

  // A fragment (16x32 bf16): lane lm = row, elements 0..7 -> K=koff..+7,
  // elements 8..15 -> K=koff+16..+23, koff = lh*8
  const __bf16* arow0 = A16 + (size_t)(mbase + lm) * D_FEAT + lh * 8;
  const __bf16* arow1 = arow0 + (size_t)16 * D_FEAT;
  // B fragment (32x16): elements 0..7 -> K=lh*8..+7, elements 8..15 -> K=lh*8+16..+23
  const __bf16* b0p = &bt[lm * LDS_STRIDE + lh * 8];
  const __bf16* b1p = b0p + 16 * LDS_STRIDE;

  v8f c00 = {}, c01 = {}, c10 = {}, c11 = {};
  for (int k = 0; k < D_FEAT; k += 32) {
    v16bf a0, a1, b0, b1;
    ((uint4*)&a0)[0] = *(const uint4*)(arow0 + k);
    ((uint4*)&a0)[1] = *(const uint4*)(arow0 + k + 16);
    ((uint4*)&a1)[0] = *(const uint4*)(arow1 + k);
    ((uint4*)&a1)[1] = *(const uint4*)(arow1 + k + 16);
    ((uint4*)&b0)[0] = *(const uint4*)(b0p + k);
    ((uint4*)&b0)[1] = *(const uint4*)(b0p + k + 16);
    ((uint4*)&b1)[0] = *(const uint4*)(b1p + k);
    ((uint4*)&b1)[1] = *(const uint4*)(b1p + k + 16);
    c00 = __builtin_amdgcn_wmma_f32_16x16x32_bf16(false, a0, false, b0, (short)0, c00, false, false);
    c01 = __builtin_amdgcn_wmma_f32_16x16x32_bf16(false, a0, false, b1, (short)0, c01, false, false);
    c10 = __builtin_amdgcn_wmma_f32_16x16x32_bf16(false, a1, false, b0, (short)0, c10, false, false);
    c11 = __builtin_amdgcn_wmma_f32_16x16x32_bf16(false, a1, false, b1, (short)0, c11, false, false);
  }

  // C/D layout: VGPR i -> M = i + 8*lh, N = lane&15
#pragma unroll
  for (int i = 0; i < 8; ++i) {
    const int m0 = mbase + lh * 8 + i;
    const size_t r0 = (size_t)m0 * N_TRAIN;
    const size_t r1 = (size_t)(m0 + 16) * N_TRAIN;
    sims[r0 + nbase + lm]      = c00[i];
    sims[r0 + nbase + 16 + lm] = c01[i];
    sims[r1 + nbase + lm]      = c10[i];
    sims[r1 + nbase + 16 + lm] = c11[i];
  }
}

// ---------------------------------------------------------------------------
// Phase 2+3: per-row radix-select top-200, bitonic sort, softmax, class scatter
// ---------------------------------------------------------------------------
__device__ __forceinline__ unsigned f2key(float v) {
  unsigned u = __float_as_uint(v);
  return (u & 0x80000000u) ? ~u : (u | 0x80000000u);  // order-preserving map
}

__global__ void __launch_bounds__(256) knn_topk_scatter(const float* __restrict__ sims,
                                                        const int* __restrict__ labels,
                                                        float* __restrict__ out) {
  const int row = blockIdx.x;
  const int tid = threadIdx.x;
  const float* srow = sims + (size_t)row * N_TRAIN;

  __shared__ unsigned hist[256];
  __shared__ unsigned s_prefix;
  __shared__ int s_need;
  __shared__ int s_cnt[2];
  __shared__ float s_val[256];
  __shared__ int   s_idx[256];
  __shared__ float s_w[256];
  __shared__ int   s_lab[256];
  __shared__ float red[256];
  __shared__ float cls[NCLS];

  if (tid == 0) { s_prefix = 0u; s_need = MAXK; }
  __syncthreads();

  // MSB-first radix select: find key of the 200th-largest element
  for (int pass = 0; pass < 4; ++pass) {
    const int shift = 24 - 8 * pass;
    const unsigned hi_mask = (pass == 0) ? 0u : (0xFFFFFFFFu << (shift + 8));
    const unsigned prefix = s_prefix;
    hist[tid] = 0u;
    __syncthreads();
    for (int i = tid; i < N_TRAIN; i += 256) {
      unsigned key = f2key(srow[i]);
      if ((key & hi_mask) == (prefix & hi_mask))
        atomicAdd(&hist[(key >> shift) & 0xFFu], 1u);
    }
    __syncthreads();
    if (tid == 0) {
      int need = s_need;
      unsigned cum = 0;
      int b;
      for (b = 255; b >= 0; --b) { cum += hist[b]; if ((int)cum >= need) break; }
      if (b < 0) b = 0;
      s_prefix = prefix | ((unsigned)b << shift);
      s_need = need - (int)(cum - hist[b]);   // ties still needed after strict-greater
    }
    __syncthreads();
  }

  const unsigned kth = s_prefix;
  if (tid == 0) { s_cnt[0] = 0; s_cnt[1] = 0; }
  __syncthreads();

  // collect strictly-greater, then enough equal-to-threshold ties
  for (int i = tid; i < N_TRAIN; i += 256) {
    float v = srow[i];
    if (f2key(v) > kth) {
      int p = atomicAdd(&s_cnt[0], 1);
      if (p < MAXK) { s_val[p] = v; s_idx[p] = i; }
    }
  }
  __syncthreads();
  const int ngt = (s_cnt[0] < MAXK) ? s_cnt[0] : MAXK;
  for (int i = tid; i < N_TRAIN; i += 256) {
    float v = srow[i];
    if (f2key(v) == kth) {
      int p = atomicAdd(&s_cnt[1], 1);
      int slot = ngt + p;
      if (slot < MAXK) { s_val[slot] = v; s_idx[slot] = i; }
    }
  }
  __syncthreads();
  if (tid >= MAXK) { s_val[tid] = -__builtin_inff(); s_idx[tid] = 0; }
  __syncthreads();

  // bitonic sort, descending, 256 elems (top-200 + -inf pad)
  for (int ksz = 2; ksz <= 256; ksz <<= 1) {
    for (int j = ksz >> 1; j > 0; j >>= 1) {
      int partner = tid ^ j;
      if (partner > tid) {
        bool desc = ((tid & ksz) == 0);
        float v0 = s_val[tid], v1 = s_val[partner];
        if (desc ? (v0 < v1) : (v0 > v1)) {
          s_val[tid] = v1; s_val[partner] = v0;
          int t = s_idx[tid]; s_idx[tid] = s_idx[partner]; s_idx[partner] = t;
        }
      }
      __syncthreads();
    }
  }

  // softmax(top-200 / T)
  const float vmax = s_val[0];
  float w = 0.f;
  if (tid < MAXK) w = __expf((s_val[tid] - vmax) * INV_TEMP);
  red[tid] = w;
  __syncthreads();
  for (int s = 128; s > 0; s >>= 1) {
    if (tid < s) red[tid] += red[tid + s];
    __syncthreads();
  }
  const float inv = 1.0f / red[0];
  if (tid < MAXK) { s_w[tid] = w * inv; s_lab[tid] = labels[s_idx[tid]]; }
  for (int c = tid; c < NCLS; c += 256) cls[c] = 0.f;
  __syncthreads();

  // incremental scatter into class bins, snapshot at each k cut
  const int cuts[4] = {10, 20, 100, 200};
  int prev = 0;
  for (int q = 0; q < 4; ++q) {
    const int cut = cuts[q];
    if (tid >= prev && tid < cut) atomicAdd(&cls[s_lab[tid]], s_w[tid]);
    __syncthreads();
    float* orow = out + ((size_t)q * B_Q + row) * NCLS;
    for (int c = tid; c < NCLS; c += 256) orow[c] = cls[c];
    __syncthreads();
    prev = cut;
  }
}

extern "C" void kernel_launch(void* const* d_in, const int* in_sizes, int n_in,
                              void* d_out, int out_size, void* d_ws, size_t ws_size,
                              hipStream_t stream) {
  (void)in_sizes; (void)n_in; (void)out_size; (void)ws_size;
  const float* feats  = (const float*)d_in[0];  // (256, 512)
  const float* train  = (const float*)d_in[1];  // (100000, 512)
  const int*   labels = (const int*)d_in[2];    // (100000,)
  float* out = (float*)d_out;                   // (4, 256, 1000)

  // workspace layout: sims f32 | train bf16 | features bf16  (~205 MB)
  const size_t SIMS_BYTES = (size_t)B_Q * N_TRAIN * sizeof(float);
  const size_t B16_BYTES  = (size_t)N_TRAIN * D_FEAT * 2;
  float*  sims = (float*)d_ws;
  __bf16* B16v = (__bf16*)((char*)d_ws + SIMS_BYTES);
  __bf16* A16v = (__bf16*)((char*)d_ws + SIMS_BYTES + B16_BYTES);

  const int nTrain8 = N_TRAIN * D_FEAT / 8;   // 6,400,000
  const int nFeat8  = B_Q * D_FEAT / 8;       // 16,384
  cvt_f32_to_bf16<<<dim3((nTrain8 + 255) / 256), dim3(256), 0, stream>>>(train, (v8bf*)B16v, nTrain8);
  cvt_f32_to_bf16<<<dim3((nFeat8 + 255) / 256), dim3(256), 0, stream>>>(feats, (v8bf*)A16v, nFeat8);
  knn_gemm_bf16<<<dim3(N_TRAIN / 32), dim3(256), 0, stream>>>(A16v, B16v, sims);
  knn_topk_scatter<<<dim3(B_Q), dim3(256), 0, stream>>>(sims, labels, out);
}
